// EpipolarCorrelationLayer_85907935855280
// MI455X (gfx1250) — matrix operand order repaired and checked by reading, not compiled, not run on hardware
//
#include <hip/hip_runtime.h>

// ---------------------------------------------------------------------------
// Epipolar correlation layer for MI455X (gfx1250).
//   out(B,85,H,W) = concat[ epipolar_flow(2), para(2), corr(81) ]
//
//  k_pack : transpose imgL/imgR (B,C,H,W) f32 -> (B,H,W,C) f16 (channel-major)
//  k_geom : per-pixel epipolar geometry; writes out ch0..3 + geo workspace
//  k_corr : per 16-pixel tile: TDM-stage a 24x24 imgR window into LDS (two
//           concurrent tensor_load_to_lds DMAs), dense f16 WMMA GEMM
//           D[576loc x 16px] = (imgR window)^T * imgL, then bilinear-blend
//           4 corner dot products per (pixel, offset).
// ---------------------------------------------------------------------------

#define BB   4
#define CC   96
#define HH   96
#define WW   320
#define WIN  24
#define NLOC (WIN * WIN)       // 576 window locations
#define PX   16                // pixels per tile (GEMM M)
#define NOFF 81                // 9x9 offsets
#define NT   (NLOC / 16)       // 36 N-tiles
#define NWAVE 4

typedef __attribute__((ext_vector_type(16))) _Float16 v16h;
typedef __attribute__((ext_vector_type(8)))  _Float16 v8h;
typedef __attribute__((ext_vector_type(8)))  float    v8f;
typedef __attribute__((ext_vector_type(4)))  float    v4f;
typedef __attribute__((ext_vector_type(4)))  unsigned v4u;
typedef __attribute__((ext_vector_type(8)))  int      v8i;
typedef __attribute__((ext_vector_type(4)))  int      v4i;

union Frag { v16h v; v8h h[2]; };
union Acc  { v8f  v; v4f q[2]; };

#if defined(__HIP_DEVICE_COMPILE__) && defined(__has_builtin)
#if __has_builtin(__builtin_amdgcn_tensor_load_to_lds)
#define HAVE_TDM 1
#endif
#endif
#ifndef HAVE_TDM
#define HAVE_TDM 0
#endif

__device__ __forceinline__ float safe_depth(float d) {
  return (fabsf(d) < 1e-6f) ? 1e-6f : d;
}

// --------------------------- pack / transpose ------------------------------
__global__ void k_pack(const float* __restrict__ L, const float* __restrict__ R,
                       _Float16* __restrict__ Lt, _Float16* __restrict__ Rt) {
  size_t i = (size_t)blockIdx.x * blockDim.x + threadIdx.x;
  size_t total = (size_t)BB * CC * HH * WW;
  if (i >= total) return;
  int x = (int)(i % WW); size_t t = i / WW;
  int y = (int)(t % HH); t /= HH;
  int c = (int)(t % CC); int b = (int)(t / CC);
  size_t o = ((((size_t)b * HH + y) * WW) + x) * (size_t)CC + c;
  Lt[o] = (_Float16)L[i];
  Rt[o] = (_Float16)R[i];
}

// ------------------------------ geometry -----------------------------------
__global__ void k_geom(const float* __restrict__ Rm, const float* __restrict__ Tm,
                       const float* __restrict__ flow, float* __restrict__ out,
                       float* __restrict__ geo) {
  const float FX = 0.89115971f * (float)WW, CX = 0.5f * (float)WW;
  const float FY = 1.18821287f * (float)HH, CY = 0.5f * (float)HH;
  int i = blockIdx.x * blockDim.x + threadIdx.x;
  const int nPix = BB * HH * WW;
  if (i >= nPix) return;
  int x = i % WW; int t = i / WW; int y = t % HH; int b = t / HH;

  float pdx = ((float)x - CX) / FX, pdy = ((float)y - CY) / FY;
  const float* Rb = Rm + b * 9;
  const float* Tb = Tm + b * 3;
  float r2  = Rb[6] * pdx + Rb[7] * pdy + Rb[8];
  float fp0 = FX * (Rb[0] * pdx + Rb[1] * pdy + Rb[2]) + CX * r2;
  float fp1 = FY * (Rb[3] * pdx + Rb[4] * pdy + Rb[5]) + CY * r2;
  float fp2 = r2;
  float s0 = FX * Tb[0] + CX * Tb[2];
  float s1 = FY * Tb[1] + CY * Tb[2];
  float s2 = Tb[2];

  float ez = safe_depth(fp2);
  float ex = fp0 / ez, ey = fp1 / ez;
  float spx = fp0 * 10.f + s0, spy = fp1 * 10.f + s1, spz = fp2 * 10.f + s2;
  float pz = safe_depth(spz);
  float prx = spx / pz, pry = spy / pz;
  float dx = prx - ex, dy = pry - ey;
  float nrm = fmaxf(sqrtf(dx * dx + dy * dy), 1e-12f);
  float pax = dx / nrm, pay = dy / nrm;

  float f0 = flow[((size_t)(b * 2 + 0) * HH + y) * WW + x];
  float f1 = flow[((size_t)(b * 2 + 1) * HH + y) * WW + x];
  float k = ((float)x + f0 - ex) * pax + ((float)y + f1 - ey) * pay;
  float ilx = ex + k * pax, ily = ey + k * pay;

  size_t ob = (size_t)b * 85;
  out[((ob + 0) * HH + y) * WW + x] = ilx - (float)x;
  out[((ob + 1) * HH + y) * WW + x] = ily - (float)y;
  out[((ob + 2) * HH + y) * WW + x] = pax;
  out[((ob + 3) * HH + y) * WW + x] = pay;
  float* g = geo + (size_t)i * 4;
  g[0] = ilx; g[1] = ily; g[2] = pax; g[3] = pay;
}

// ---------------------------- correlation ----------------------------------
#define SM_B_HALVES (NLOC * CC)                       // 55296 f16 = 110592 B
#define SMEM_BYTES  (SM_B_HALVES * 2 + NLOC * PX * 4) // + 36864 B for D

#if HAVE_TDM
__device__ __forceinline__ void tdm_load_rows(const _Float16* gsrc, void* ldst,
                                              int nrows) {
  unsigned laddr = (unsigned)(unsigned long long)ldst;
  unsigned long long gaddr = (unsigned long long)(const void*)gsrc;
  v4u g0;
  g0[0] = 1u;                                          // count=1 valid descriptor
  g0[1] = laddr;                                       // lds_addr
  g0[2] = (unsigned)gaddr;                             // global_addr lo
  g0[3] = (unsigned)((gaddr >> 32) & 0x1FFFFFFull) | 0x80000000u; // hi | type=2
  v8i g1;
  g1[0] = 1 << 16;                                     // data_size = 2B
  g1[1] = (int)(((unsigned)(WW * CC) & 0xFFFFu) << 16);            // dim0 lo
  g1[2] = (int)(((unsigned)(WW * CC) >> 16) & 0xFFFFu)
        | (int)(((unsigned)HH & 0xFFFFu) << 16);                   // dim0 hi|dim1 lo
  g1[3] = (int)(((unsigned)(WIN * CC)) << 16);                     // tile_dim0=2304
  g1[4] = nrows;                                       // tile_dim1 lines
  g1[5] = WW * CC;                                     // dim0 stride (elems)
  g1[6] = 0;
  g1[7] = 0;
  v4i gz = {0, 0, 0, 0};
#if __clang_major__ >= 23
  v8i gz8 = {0, 0, 0, 0, 0, 0, 0, 0};
  __builtin_amdgcn_tensor_load_to_lds(g0, g1, gz, gz, gz8, 0);
#else
  __builtin_amdgcn_tensor_load_to_lds(g0, g1, gz, gz, 0);
#endif
}
#endif

__global__ __launch_bounds__(NWAVE * 32) void
k_corr(const _Float16* __restrict__ Lt, const _Float16* __restrict__ Rt,
       const float* __restrict__ geo, float* __restrict__ out) {
  extern __shared__ __align__(16) char smem[];
  _Float16* bwin = (_Float16*)smem;                    // window [loc][c] f16
  float*    Dl   = (float*)(smem + SM_B_HALVES * 2);   // D[loc][pixel] f32

  const int x0   = blockIdx.x * PX;
  const int y    = blockIdx.y;
  const int b    = blockIdx.z;
  const int tid  = threadIdx.x;
  const int lane = tid & 31;
  const int wave = tid >> 5;

  // ---- wave-uniform window origin from tile-center geometry ----
  size_t pc = (((size_t)b * HH + y) * WW + x0 + PX / 2);
  float cilx = geo[pc * 4 + 0], cily = geo[pc * 4 + 1];
  float cpax = geo[pc * 4 + 2], cpay = geo[pc * 4 + 3];
  int ox = (int)floorf(cilx - cpay) - (WIN / 2 - 1);   // center ~= il + perp
  int oy = (int)floorf(cily + cpax) - (WIN / 2 - 1);
  ox = min(max(ox, 0), WW - WIN);
  oy = min(max(oy, 0), HH - WIN);
  ox = __builtin_amdgcn_readfirstlane(ox);
  oy = __builtin_amdgcn_readfirstlane(oy);

  const _Float16* wsrc = Rt + ((((size_t)b * HH + oy) * WW) + ox) * (size_t)CC;

  // ---- stage 24x24x96 f16 window into LDS (two concurrent TDM DMAs) ----
#if HAVE_TDM
  if (wave < 2) {
    const int half = WIN / 2;                 // 12 rows per DMA
    tdm_load_rows(wsrc + (size_t)wave * half * WW * CC,
                  (char*)bwin + (size_t)wave * half * WIN * CC * 2, half);
  }
#else
  for (int c = tid; c < SM_B_HALVES / 8; c += NWAVE * 32) {
    int wy  = c / (WIN * CC / 8);
    int rem = c - wy * (WIN * CC / 8);
    ((v8h*)bwin)[c] = *(const v8h*)(wsrc + (size_t)wy * WW * CC + rem * 8);
  }
#endif

  // ---- A fragments (overlap global latency with the tensor DMA) ----
  const int mrow = lane & 15;
  const _Float16* ap = Lt + ((((size_t)b * HH + y) * WW) + x0 + mrow) * (size_t)CC;
  Frag afr[3];
#pragma unroll
  for (int s = 0; s < 3; ++s) {
    int k0 = s * 32 + ((lane < 16) ? 0 : 8);
    int k1 = s * 32 + ((lane < 16) ? 16 : 24);
    afr[s].h[0] = *(const v8h*)(ap + k0);
    afr[s].h[1] = *(const v8h*)(ap + k1);
  }

#if HAVE_TDM
  if (wave < 2) {
#if __has_builtin(__builtin_amdgcn_s_wait_tensorcnt)
    __builtin_amdgcn_s_wait_tensorcnt(0);
#else
    asm volatile("s_wait_tensorcnt 0x0" ::: "memory");
#endif
  }
#endif
  __syncthreads();

  // ---- GEMM: D[576 x 16] = B^T * A, 4 waves x 2 tiles/iter (2 acc chains) --
  for (int base = wave * 2; base < NT; base += NWAVE * 2) {
    const int nt0 = base, nt1 = base + 1;
    const _Float16* bp0 = bwin + (size_t)(nt0 * 16 + (lane & 15)) * CC;
    const _Float16* bp1 = bwin + (size_t)(nt1 * 16 + (lane & 15)) * CC;
    Acc acc0, acc1;
    acc0.v = (v8f){0.f, 0.f, 0.f, 0.f, 0.f, 0.f, 0.f, 0.f};
    acc1.v = acc0.v;
#pragma unroll
    for (int s = 0; s < 3; ++s) {
      int k0 = s * 32 + ((lane < 16) ? 0 : 8);
      Frag bf0, bf1;
      bf0.h[0] = *(const v8h*)(bp0 + k0);
      bf0.h[1] = *(const v8h*)(bp0 + k0 + 16);
      bf1.h[0] = *(const v8h*)(bp1 + k0);
      bf1.h[1] = *(const v8h*)(bp1 + k0 + 16);
      acc0.v = __builtin_amdgcn_wmma_f32_16x16x32_f16(
          false, afr[s].v, false, bf0.v, (short)0, acc0.v, false, false);
      acc1.v = __builtin_amdgcn_wmma_f32_16x16x32_f16(
          false, afr[s].v, false, bf1.v, (short)0, acc1.v, false, false);
    }
    // lane holds D column N (= window loc), rows M=Mb..Mb+7 -> contiguous
    int N = lane & 15, Mb = (lane >> 4) * 8;
    float* d0 = Dl + (size_t)(nt0 * 16 + N) * PX + Mb;
    float* d1 = Dl + (size_t)(nt1 * 16 + N) * PX + Mb;
    *(v4f*)(d0)     = acc0.q[0];
    *(v4f*)(d0 + 4) = acc0.q[1];
    *(v4f*)(d1)     = acc1.q[0];
    *(v4f*)(d1 + 4) = acc1.q[1];
  }
  __syncthreads();

  // ---- phase 2: bilinear blend of corner dot-products per (pixel, offset) --
  // lanes 0..15 share an offset o and span the 16 pixels -> coalesced stores.
  const float WSX = (float)WW / (float)(WW - 1);
  const float WSY = (float)HH / (float)(HH - 1);
  for (int t = tid; t < PX * NOFF; t += NWAVE * 32) {
    int p = t & 15;
    int o = t >> 4;
    float a = (float)(o / 9) - 4.0f;   // para coefficient
    float q = (float)(o % 9) - 4.0f;   // scalar perp_i added to both comps
    size_t pix = (((size_t)b * HH + y) * WW + x0 + p);
    const float* g4 = geo + pix * 4;
    float ilx = g4[0], ily = g4[1], pax = g4[2], pay = g4[3];
    // g = initial_loc + a*para + q + perp ; perp = (-pay, pax)
    float gx_ = ilx + a * pax + q - pay;
    float gy_ = ily + a * pay + q + pax;
    // grid_sample coord round-trip: s = g * N/(N-1) - 0.5
    float sx = gx_ * WSX - 0.5f;
    float sy = gy_ * WSY - 0.5f;
    float fx0 = floorf(sx), fy0 = floorf(sy);
    float wx = sx - fx0, wy = sy - fy0;
    int ix0 = (int)fx0, iy0 = (int)fy0;
    float corr = 0.f;
#pragma unroll
    for (int dyy = 0; dyy < 2; ++dyy) {
#pragma unroll
      for (int dxx = 0; dxx < 2; ++dxx) {
        int xi = ix0 + dxx, yi = iy0 + dyy;
        float w = (dxx ? wx : 1.f - wx) * (dyy ? wy : 1.f - wy);
        int lx = xi - ox, ly = yi - oy;
        bool ok = (xi >= 0) & (xi < WW) & (yi >= 0) & (yi < HH) &
                  (lx >= 0) & (lx < WIN) & (ly >= 0) & (ly < WIN);
        float v = ok ? Dl[(size_t)(ly * WIN + lx) * PX + p] : 0.f;
        corr += w * v;
      }
    }
    out[(((size_t)b * 85 + 4 + o) * HH + y) * WW + (x0 + p)] = corr * (1.0f / CC);
  }
}

// ------------------------------- launcher ----------------------------------
extern "C" void kernel_launch(void* const* d_in, const int* in_sizes, int n_in,
                              void* d_out, int out_size, void* d_ws, size_t ws_size,
                              hipStream_t stream) {
  const float* imgL = (const float*)d_in[0];
  const float* imgR = (const float*)d_in[1];
  const float* Rm   = (const float*)d_in[2];
  const float* Tm   = (const float*)d_in[3];
  const float* flow = (const float*)d_in[4];
  float* out = (float*)d_out;

  size_t nPix = (size_t)BB * HH * WW;
  float*    geo = (float*)d_ws;                                   // 4 f32 / pixel
  _Float16* Lt  = (_Float16*)((char*)d_ws + nPix * 4 * sizeof(float));
  _Float16* Rt  = Lt + nPix * CC;

  size_t total = (size_t)BB * CC * HH * WW;
  k_pack<<<(unsigned)((total + 255) / 256), 256, 0, stream>>>(imgL, imgR, Lt, Rt);
  k_geom<<<(unsigned)((nPix + 255) / 256), 256, 0, stream>>>(Rm, Tm, flow, out, geo);

  dim3 grid(WW / PX, HH, BB);   // 20 x 96 x 4 = 7680 tiles
  k_corr<<<grid, NWAVE * 32, (size_t)SMEM_BYTES, stream>>>(Lt, Rt, geo, out);

  (void)in_sizes; (void)n_in; (void)out_size; (void)ws_size;
}
